// Proposed_47957604827359
// MI455X (gfx1250) — matrix operand order repaired and checked by reading, not compiled, not run on hardware
//
#include <hip/hip_runtime.h>

#define N_NODES 50000
#define D_IN    512
#define HID     256
#define HEADS   4
#define OUT_D   64
#define MLP_H   128
#define L_LAYERS 2
#define QKV     (HID * HEADS)   // 1024

typedef __bf16 bf16_t;
typedef __attribute__((ext_vector_type(16))) __bf16 v16bf;
typedef __attribute__((ext_vector_type(8)))  __bf16 v8bf;
typedef __attribute__((ext_vector_type(8)))  float  v8f;
typedef __attribute__((ext_vector_type(8)))  float  v8fv;

#define WMMA_BF16(acc, a, b) \
  __builtin_amdgcn_wmma_f32_16x16x32_bf16(false, (a), false, (b), (short)0, (acc), false, false)

// Load a 16-element bf16 fragment as two 16B vector loads (global or LDS).
__device__ __forceinline__ v16bf ld_frag(const bf16_t* p, int hioff) {
  v8bf lo = *(const v8bf*)p;
  v8bf hi = *(const v8bf*)(p + hioff);
  return __builtin_shufflevector(lo, hi, 0, 1, 2, 3, 4, 5, 6, 7,
                                 8, 9, 10, 11, 12, 13, 14, 15);
}

// ---------------------------------------------------------------------------
// LDS-free bf16 GEMM: C[M,Nc] = A[M,K] @ Bt[Nc,K]^T + bias (+relu)
// Software-pipelined: next K-step's 5 fragments load while current 4 wmma run.
// ---------------------------------------------------------------------------
__global__ __launch_bounds__(256) void bf16_gemm_kernel(
    const bf16_t* __restrict__ A, int lda,
    const bf16_t* __restrict__ Bt, int ldk,
    const float* __restrict__ bias,
    float* __restrict__ C, bf16_t* __restrict__ Cb, int ldc,
    int M, int Kdim, int relu)
{
  const int lane = threadIdx.x & 31, w = threadIdx.x >> 5;
  const int m0 = blockIdx.x * 128 + w * 16;
  const int n0 = blockIdx.y * 64;

  int arow = m0 + (lane & 15);
  if (arow >= M) arow = M - 1;                     // clamp: OOB rows never stored
  const bf16_t* Ap  = A + (long)arow * lda + (lane >> 4) * 16;
  const bf16_t* Bp0 = Bt + (long)(n0 +  0 + (lane & 15)) * ldk + (lane >> 4) * 16;
  const bf16_t* Bp1 = Bt + (long)(n0 + 16 + (lane & 15)) * ldk + (lane >> 4) * 16;
  const bf16_t* Bp2 = Bt + (long)(n0 + 32 + (lane & 15)) * ldk + (lane >> 4) * 16;
  const bf16_t* Bp3 = Bt + (long)(n0 + 48 + (lane & 15)) * ldk + (lane >> 4) * 16;

  v8f vz = {};
  v8f acc0 = vz, acc1 = vz, acc2 = vz, acc3 = vz;

  // prologue loads
  v16bf a  = ld_frag(Ap, 8);
  v16bf b0 = ld_frag(Bp0, 8);
  v16bf b1 = ld_frag(Bp1, 8);
  v16bf b2 = ld_frag(Bp2, 8);
  v16bf b3 = ld_frag(Bp3, 8);

  for (int kk0 = 32; kk0 < Kdim; kk0 += 32) {
    v16bf an = ld_frag(Ap + kk0, 8);               // prefetch next step
    v16bf c0 = ld_frag(Bp0 + kk0, 8);
    v16bf c1 = ld_frag(Bp1 + kk0, 8);
    v16bf c2 = ld_frag(Bp2 + kk0, 8);
    v16bf c3 = ld_frag(Bp3 + kk0, 8);
    acc0 = WMMA_BF16(acc0, a, b0);
    acc1 = WMMA_BF16(acc1, a, b1);
    acc2 = WMMA_BF16(acc2, a, b2);
    acc3 = WMMA_BF16(acc3, a, b3);
    a = an; b0 = c0; b1 = c1; b2 = c2; b3 = c3;
  }
  acc0 = WMMA_BF16(acc0, a, b0);
  acc1 = WMMA_BF16(acc1, a, b1);
  acc2 = WMMA_BF16(acc2, a, b2);
  acc3 = WMMA_BF16(acc3, a, b3);

  const int hs = (lane >> 4) * 8;                  // C/D: VGPR r -> M = r + 8*(lane>=16)
  v8f accs[4] = {acc0, acc1, acc2, acc3};
#pragma unroll
  for (int t = 0; t < 4; ++t) {
    int gc = n0 + t * 16 + (lane & 15);
    float bv = bias[gc];
#pragma unroll
    for (int r = 0; r < 8; ++r) {
      int gr = m0 + r + hs;
      if (gr < M) {
        float v = accs[t][r] + bv;
        if (relu) v = fmaxf(v, 0.f);
        if (C)  C[(long)gr * ldc + gc] = v;
        if (Cb) Cb[(long)gr * ldc + gc] = (bf16_t)v;
      }
    }
  }
}

// ---------------------------------------------------------------------------
// kvsT[h,d,m] += sum_n k[n,h*256+m] * v[n,h*256+d]   (split-K, f32 atomics)
// Both operands need the n-transpose -> LDS staged. Stores transposed result.
// ---------------------------------------------------------------------------
__global__ __launch_bounds__(256) void kvs_kernel(
    const bf16_t* __restrict__ Kb, const bf16_t* __restrict__ Vb,
    float* __restrict__ kvsT, int nrows, int chunkSpan, int nChunks)
{
  __shared__ bf16_t sA[64 * 32];                   // [m][n-within-step]
  __shared__ bf16_t sBt[64 * 32];                  // [d][n-within-step]
  const int tid = threadIdx.x, lane = tid & 31, w = tid >> 5;
  const int m0 = blockIdx.x * 64, d0 = blockIdx.y * 64;
  const int head = blockIdx.z / nChunks, chunk = blockIdx.z % nChunks;
  const int wr = (w & 3) * 16, wc = (w >> 2) * 32;
  v8f vz = {};
  v8f acc0 = vz, acc1 = vz;

  const int c  = tid >> 3;                         // n index within step (0..31)
  const int m8 = (tid & 7) * 8;                    // element base (0..56)
  const int nbase0 = chunk * chunkSpan;

  for (int s = 0; s < chunkSpan; s += 32) {
    int nb = nbase0 + s;
    if (nb >= nrows) break;                        // uniform across block
    int n = nb + c;
    bool ok = (n < nrows);
    if (!ok) n = nrows - 1;                        // clamped, branchless
    v8bf ka = *(const v8bf*)(Kb + (long)n * QKV + head * HID + m0 + m8);
    v8bf va = *(const v8bf*)(Vb + (long)n * QKV + head * HID + d0 + m8);
#pragma unroll
    for (int j = 0; j < 8; ++j) {
      sA[(m8 + j) * 32 + c]  = ok ? ka[j] : (bf16_t)0.f;
      sBt[(m8 + j) * 32 + c] = ok ? va[j] : (bf16_t)0.f;
    }
    __syncthreads();
    v16bf a  = ld_frag(sA + (wr + (lane & 15)) * 32 + (lane >> 4) * 16, 8);
    v16bf b0 = ld_frag(sBt + (wc + (lane & 15)) * 32 + (lane >> 4) * 16, 8);
    v16bf b1 = ld_frag(sBt + (wc + 16 + (lane & 15)) * 32 + (lane >> 4) * 16, 8);
    acc0 = WMMA_BF16(acc0, a, b0);
    acc1 = WMMA_BF16(acc1, a, b1);
    __syncthreads();
  }

  const int hs = (lane >> 4) * 8;
#pragma unroll
  for (int r = 0; r < 8; ++r) {
    int gm = m0 + wr + r + hs;
    int gd = d0 + wc + (lane & 15);
    atomicAdd(&kvsT[((long)head * HID + gd) * HID + gm],      acc0[r]);
    atomicAdd(&kvsT[((long)head * HID + gd + 16) * HID + gm], acc1[r]);
  }
}

// ---------------------------------------------------------------------------
// out[n,d] = 0.25 * sum_h ( scale*(q_h[n,:]@kvs[h,:,d]) + M*v[n,h,d] ) / denom[n,h]
// LDS-free, software-pipelined per head (K = 256 = 8 steps).
// ---------------------------------------------------------------------------
__global__ __launch_bounds__(256) void attn_kernel(
    const bf16_t* __restrict__ Qb, const bf16_t* __restrict__ kvsTb,
    const bf16_t* __restrict__ Vb, const float* __restrict__ denom,
    const float* __restrict__ scal, float* __restrict__ outp, int M)
{
  const int lane = threadIdx.x & 31, w = threadIdx.x >> 5;
  const int m0 = blockIdx.x * 128 + w * 16;
  const int d0 = blockIdx.y * 64;
  const float scale = rsqrtf(scal[0] * scal[1]);
  const int hs = (lane >> 4) * 8;

  int arow = m0 + (lane & 15);
  if (arow >= M) arow = M - 1;

  float run[4][8];
#pragma unroll
  for (int t = 0; t < 4; ++t)
#pragma unroll
    for (int r = 0; r < 8; ++r) run[t][r] = 0.f;

  for (int head = 0; head < HEADS; ++head) {
    const bf16_t* Ap = Qb + (long)arow * QKV + head * HID + (lane >> 4) * 16;
    const bf16_t* Bbase = kvsTb + (long)head * HID * HID;
    const bf16_t* Bp0 = Bbase + (long)(d0 +  0 + (lane & 15)) * HID + (lane >> 4) * 16;
    const bf16_t* Bp1 = Bbase + (long)(d0 + 16 + (lane & 15)) * HID + (lane >> 4) * 16;
    const bf16_t* Bp2 = Bbase + (long)(d0 + 32 + (lane & 15)) * HID + (lane >> 4) * 16;
    const bf16_t* Bp3 = Bbase + (long)(d0 + 48 + (lane & 15)) * HID + (lane >> 4) * 16;

    v8f vz = {};
    v8f acc0 = vz, acc1 = vz, acc2 = vz, acc3 = vz;

    v16bf a  = ld_frag(Ap, 8);
    v16bf b0 = ld_frag(Bp0, 8);
    v16bf b1 = ld_frag(Bp1, 8);
    v16bf b2 = ld_frag(Bp2, 8);
    v16bf b3 = ld_frag(Bp3, 8);
    for (int kk0 = 32; kk0 < HID; kk0 += 32) {
      v16bf an = ld_frag(Ap + kk0, 8);
      v16bf c0 = ld_frag(Bp0 + kk0, 8);
      v16bf c1 = ld_frag(Bp1 + kk0, 8);
      v16bf c2 = ld_frag(Bp2 + kk0, 8);
      v16bf c3 = ld_frag(Bp3 + kk0, 8);
      acc0 = WMMA_BF16(acc0, a, b0);
      acc1 = WMMA_BF16(acc1, a, b1);
      acc2 = WMMA_BF16(acc2, a, b2);
      acc3 = WMMA_BF16(acc3, a, b3);
      a = an; b0 = c0; b1 = c1; b2 = c2; b3 = c3;
    }
    acc0 = WMMA_BF16(acc0, a, b0);
    acc1 = WMMA_BF16(acc1, a, b1);
    acc2 = WMMA_BF16(acc2, a, b2);
    acc3 = WMMA_BF16(acc3, a, b3);

    v8f accs[4] = {acc0, acc1, acc2, acc3};
#pragma unroll
    for (int t = 0; t < 4; ++t) {
      int gc = d0 + t * 16 + (lane & 15);
#pragma unroll
      for (int r = 0; r < 8; ++r) {
        int gr = m0 + r + hs;
        if (gr < M) {
          float den = denom[gr * HEADS + head];
          float num = scale * accs[t][r] +
                      (float)M * (float)Vb[(long)gr * QKV + head * HID + gc];
          run[t][r] += num / den;
        }
      }
    }
  }
#pragma unroll
  for (int t = 0; t < 4; ++t) {
    int gc = d0 + t * 16 + (lane & 15);
#pragma unroll
    for (int r = 0; r < 8; ++r) {
      int gr = m0 + r + hs;
      if (gr < M) outp[(long)gr * HID + gc] = run[t][r] * 0.25f;
    }
  }
}

// ------------------------------ small kernels ------------------------------

__global__ __launch_bounds__(256) void zero_kernel(float* __restrict__ p, int n) {
  for (int i = blockIdx.x * 256 + threadIdx.x; i < n; i += gridDim.x * 256) p[i] = 0.f;
}

__global__ __launch_bounds__(256) void cvt_bf16_kernel(
    const float* __restrict__ in, bf16_t* __restrict__ outp, long n8)
{
  long stride = (long)gridDim.x * 256;
  for (long i = blockIdx.x * 256L + threadIdx.x; i < n8; i += stride) {
    v8fv v = *(const v8fv*)(in + i * 8);
    v8bf o;
#pragma unroll
    for (int j = 0; j < 8; ++j) o[j] = (bf16_t)v[j];
    *(v8bf*)(outp + i * 8) = o;
  }
}

// weights: in[K][N] fp32 -> out[N][K] bf16 (write-contiguous)
__global__ __launch_bounds__(256) void transpose_cvt_kernel(
    const float* __restrict__ in, bf16_t* __restrict__ outp, int K, int N)
{
  int idx = blockIdx.x * 256 + threadIdx.x;
  if (idx >= K * N) return;
  int n = idx / K, k = idx - n * K;
  outp[idx] = (bf16_t)in[(long)k * N + n];
}

__global__ __launch_bounds__(256) void sumsq_bf16_kernel(
    const bf16_t* __restrict__ p, long n, float* __restrict__ outp)
{
  float s = 0.f;
  long stride = (long)gridDim.x * 256 * 8;
  for (long i = (blockIdx.x * 256L + threadIdx.x) * 8; i < n; i += stride) {
    v8bf v = *(const v8bf*)(p + i);
#pragma unroll
    for (int j = 0; j < 8; ++j) { float f = (float)v[j]; s += f * f; }
  }
  for (int o = 16; o > 0; o >>= 1) s += __shfl_down(s, o, 32);
  __shared__ float red[8];
  if ((threadIdx.x & 31) == 0) red[threadIdx.x >> 5] = s;
  __syncthreads();
  if (threadIdx.x == 0) {
    float t = 0.f;
    for (int i = 0; i < 8; ++i) t += red[i];
    atomicAdd(outp, t);
  }
}

__global__ __launch_bounds__(256) void colsum_bf16_kernel(
    const bf16_t* __restrict__ Kb, float* __restrict__ ks, int nrows, int rowsPerBlock)
{
  int col = blockIdx.x * 256 + threadIdx.x;
  int r0 = blockIdx.y * rowsPerBlock;
  int r1 = r0 + rowsPerBlock; if (r1 > nrows) r1 = nrows;
  float s = 0.f;
  for (int r = r0; r < r1; ++r) s += (float)Kb[(long)r * QKV + col];
  atomicAdd(&ks[col], s);
}

__global__ __launch_bounds__(256) void denom_kernel(
    const bf16_t* __restrict__ Qb, const float* __restrict__ ks,
    const float* __restrict__ scal, float* __restrict__ den, int M)
{
  int i = blockIdx.x * 256 + threadIdx.x;
  if (i >= M * HEADS) return;
  int n = i >> 2, h = i & 3;
  float scale = rsqrtf(scal[0] * scal[1]);
  const bf16_t* qp = Qb + (long)n * QKV + h * HID;
  const float* wp = ks + h * HID;
  float s = 0.f;
  for (int m = 0; m < HID; m += 8) {
    v8bf qv = *(const v8bf*)(qp + m);
#pragma unroll
    for (int j = 0; j < 8; ++j) s += (float)qv[j] * wp[m + j];
  }
  den[i] = scale * s + (float)M;
}

__global__ __launch_bounds__(256) void ln_relu_kernel(
    const float* __restrict__ inp, const float* __restrict__ prev,
    const float* __restrict__ gw, const float* __restrict__ gb,
    float* __restrict__ outp, bf16_t* __restrict__ outb, int M)
{
  int n = blockIdx.x, t = threadIdx.x;
  float v = inp[(long)n * HID + t];
  if (prev) v = 0.5f * (v + prev[(long)n * HID + t]);
  __shared__ float red[8];
  float s = v;
  for (int o = 16; o > 0; o >>= 1) s += __shfl_down(s, o, 32);
  if ((t & 31) == 0) red[t >> 5] = s;
  __syncthreads();
  float mu = 0.f; for (int i = 0; i < 8; ++i) mu += red[i];
  mu *= (1.f / HID);
  __syncthreads();
  float d = v - mu;
  s = d * d;
  for (int o = 16; o > 0; o >>= 1) s += __shfl_down(s, o, 32);
  if ((t & 31) == 0) red[t >> 5] = s;
  __syncthreads();
  float var = 0.f; for (int i = 0; i < 8; ++i) var += red[i];
  var *= (1.f / HID);
  float r = d * rsqrtf(var + 1e-5f) * gw[t] + gb[t];
  r = fmaxf(r, 0.f);
  outp[(long)n * HID + t] = r;
  outb[(long)n * HID + t] = (bf16_t)r;
}

__global__ __launch_bounds__(256) void lsm_combine_kernel(
    const float* __restrict__ gpre, const float* __restrict__ x1,
    bf16_t* __restrict__ outb, int M)
{
  int n = blockIdx.x, t = threadIdx.x;
  float v = gpre[(long)n * HID + t];
  __shared__ float red[8];
  float s = v;
  for (int o = 16; o > 0; o >>= 1) s = fmaxf(s, __shfl_down(s, o, 32));
  if ((t & 31) == 0) red[t >> 5] = s;
  __syncthreads();
  float mx = red[0]; for (int i = 1; i < 8; ++i) mx = fmaxf(mx, red[i]);
  __syncthreads();
  s = __expf(v - mx);
  for (int o = 16; o > 0; o >>= 1) s += __shfl_down(s, o, 32);
  if ((t & 31) == 0) red[t >> 5] = s;
  __syncthreads();
  float se = 0.f; for (int i = 0; i < 8; ++i) se += red[i];
  float lsm = (v - mx) - __logf(se);
  outb[(long)n * HID + t] = (bf16_t)(0.5f * lsm + 0.5f * x1[(long)n * HID + t]);
}

// ------------------------------- launcher ---------------------------------

extern "C" void kernel_launch(void* const* d_in, const int* in_sizes, int n_in,
                              void* d_out, int out_size, void* d_ws, size_t ws_size,
                              hipStream_t stream) {
  (void)in_sizes; (void)n_in; (void)out_size; (void)ws_size;
  const float* x         = (const float*)d_in[0];
  /* d_in[1] edge_index unused by the reference */
  const float* xs        = (const float*)d_in[2];
  const float* fc0_w     = (const float*)d_in[3];
  const float* fc0_b     = (const float*)d_in[4];
  const float* ln_w      = (const float*)d_in[5];
  const float* ln_b      = (const float*)d_in[6];
  const float* Wq_w      = (const float*)d_in[7];
  const float* Wq_b      = (const float*)d_in[8];
  const float* Wk_w      = (const float*)d_in[9];
  const float* Wk_b      = (const float*)d_in[10];
  const float* Wv_w      = (const float*)d_in[11];
  const float* Wv_b      = (const float*)d_in[12];
  const float* mlp_w     = (const float*)d_in[13];
  const float* mlp_b     = (const float*)d_in[14];
  const float* mlp_lin_w = (const float*)d_in[15];
  const float* mlp_lin_b = (const float*)d_in[16];
  const float* fc_w      = (const float*)d_in[17];
  const float* fc_b      = (const float*)d_in[18];
  float* out = (float*)d_out;

  char* wsb = (char*)d_ws;
  size_t off = 0;
  auto takeF = [&](size_t nelem) {
    float* p = (float*)(wsb + off);
    off += ((nelem * 4 + 255) & ~(size_t)255);
    return p;
  };
  auto takeB = [&](size_t nelem) {
    bf16_t* p = (bf16_t*)(wsb + off);
    off += ((nelem * 2 + 255) & ~(size_t)255);
    return p;
  };
  float* tmp   = takeF((size_t)N_NODES * HID);
  float* h0    = takeF((size_t)N_NODES * HID);
  float* h1    = takeF((size_t)N_NODES * HID);
  float* kvsT  = takeF((size_t)HEADS * HID * HID);
  float* kssum = takeF(QKV);
  float* scal  = takeF(64);
  float* den   = takeF((size_t)N_NODES * HEADS);
  bf16_t* xb    = takeB((size_t)N_NODES * D_IN);
  bf16_t* xsb   = takeB((size_t)4 * N_NODES * D_IN);
  bf16_t* h0b   = takeB((size_t)N_NODES * HID);
  bf16_t* h1b   = takeB((size_t)N_NODES * HID);
  bf16_t* qb    = takeB((size_t)N_NODES * QKV);
  bf16_t* kbb   = takeB((size_t)N_NODES * QKV);
  bf16_t* vbb   = takeB((size_t)N_NODES * QKV);
  bf16_t* kvsTb = takeB((size_t)HEADS * HID * HID);
  bf16_t* wfc0t = takeB((size_t)HID * D_IN);
  bf16_t* wqt   = takeB((size_t)L_LAYERS * QKV * HID);
  bf16_t* wkt   = takeB((size_t)L_LAYERS * QKV * HID);
  bf16_t* wvt   = takeB((size_t)L_LAYERS * QKV * HID);
  bf16_t* wmlpt = takeB((size_t)4 * MLP_H * D_IN);
  bf16_t* wlint = takeB((size_t)HID * 4 * MLP_H);
  bf16_t* wfct  = takeB((size_t)OUT_D * HID);

  const dim3 blk(256);
  const int MT = (N_NODES + 127) / 128;
  const int CH = 4096;
  const int nChunks = (N_NODES + CH - 1) / CH;

  // ---- conversions ----
  cvt_bf16_kernel<<<dim3(2048), blk, 0, stream>>>(x,  xb,  (long)N_NODES * D_IN / 8);
  cvt_bf16_kernel<<<dim3(4096), blk, 0, stream>>>(xs, xsb, (long)4 * N_NODES * D_IN / 8);
  transpose_cvt_kernel<<<dim3((D_IN * HID + 255) / 256), blk, 0, stream>>>(fc0_w, wfc0t, D_IN, HID);
  for (int i = 0; i < L_LAYERS; ++i) {
    size_t wo = (size_t)i * HID * QKV;
    transpose_cvt_kernel<<<dim3((HID * QKV + 255) / 256), blk, 0, stream>>>(Wq_w + wo, wqt + wo, HID, QKV);
    transpose_cvt_kernel<<<dim3((HID * QKV + 255) / 256), blk, 0, stream>>>(Wk_w + wo, wkt + wo, HID, QKV);
    transpose_cvt_kernel<<<dim3((HID * QKV + 255) / 256), blk, 0, stream>>>(Wv_w + wo, wvt + wo, HID, QKV);
  }
  for (int kk = 0; kk < 4; ++kk)
    transpose_cvt_kernel<<<dim3((D_IN * MLP_H + 255) / 256), blk, 0, stream>>>(
        mlp_w + (size_t)kk * D_IN * MLP_H, wmlpt + (size_t)kk * MLP_H * D_IN, D_IN, MLP_H);
  transpose_cvt_kernel<<<dim3((4 * MLP_H * HID + 255) / 256), blk, 0, stream>>>(mlp_lin_w, wlint, 4 * MLP_H, HID);
  transpose_cvt_kernel<<<dim3((HID * OUT_D + 255) / 256), blk, 0, stream>>>(fc_w, wfct, HID, OUT_D);

  // ---- fc0 + LN + relu ----
  bf16_gemm_kernel<<<dim3(MT, HID / 64), blk, 0, stream>>>(
      xb, D_IN, wfc0t, D_IN, fc0_b, tmp, nullptr, HID, N_NODES, D_IN, 0);
  ln_relu_kernel<<<dim3(N_NODES), blk, 0, stream>>>(tmp, nullptr, ln_w, ln_b, h0, h0b, N_NODES);

  float* hc = h0; float* hn = h1;
  bf16_t* hcb = h0b; bf16_t* hnb = h1b;
  for (int i = 0; i < L_LAYERS; ++i) {
    size_t wo = (size_t)i * HID * QKV;
    bf16_gemm_kernel<<<dim3(MT, QKV / 64), blk, 0, stream>>>(
        hcb, HID, wqt + wo, HID, Wq_b + i * QKV, nullptr, qb, QKV, N_NODES, HID, 0);
    bf16_gemm_kernel<<<dim3(MT, QKV / 64), blk, 0, stream>>>(
        hcb, HID, wkt + wo, HID, Wk_b + i * QKV, nullptr, kbb, QKV, N_NODES, HID, 0);
    bf16_gemm_kernel<<<dim3(MT, QKV / 64), blk, 0, stream>>>(
        hcb, HID, wvt + wo, HID, Wv_b + i * QKV, nullptr, vbb, QKV, N_NODES, HID, 0);
    int zc = HEADS * HID * HID + QKV + 64;
    zero_kernel<<<dim3((zc + 255) / 256), blk, 0, stream>>>(kvsT, zc);
    sumsq_bf16_kernel<<<dim3(1024), blk, 0, stream>>>(qb,  (long)N_NODES * QKV, scal + 0);
    sumsq_bf16_kernel<<<dim3(1024), blk, 0, stream>>>(kbb, (long)N_NODES * QKV, scal + 1);
    colsum_bf16_kernel<<<dim3(QKV / 256, nChunks), blk, 0, stream>>>(kbb, kssum, N_NODES, CH);
    kvs_kernel<<<dim3(HID / 64, HID / 64, HEADS * nChunks), blk, 0, stream>>>(
        kbb, vbb, kvsT, N_NODES, CH, nChunks);
    cvt_bf16_kernel<<<dim3(128), blk, 0, stream>>>(kvsT, kvsTb, (long)HEADS * HID * HID / 8);
    denom_kernel<<<dim3((N_NODES * HEADS + 255) / 256), blk, 0, stream>>>(
        qb, kssum, scal, den, N_NODES);
    attn_kernel<<<dim3(MT, HID / 64), blk, 0, stream>>>(qb, kvsTb, vbb, den, scal, tmp, N_NODES);
    ln_relu_kernel<<<dim3(N_NODES), blk, 0, stream>>>(
        tmp, hc, ln_w + (i + 1) * HID, ln_b + (i + 1) * HID, hn, hnb, N_NODES);
    float* sf = hc; hc = hn; hn = sf;
    bf16_t* sb = hcb; hcb = hnb; hnb = sb;
  }
  float* x1 = hc;

  // ---- SIGN branch ----
  bf16_t* featb = qb;
  for (int kk = 0; kk < 4; ++kk) {
    bf16_gemm_kernel<<<dim3(MT, MLP_H / 64), blk, 0, stream>>>(
        xsb + (size_t)kk * N_NODES * D_IN, D_IN,
        wmlpt + (size_t)kk * MLP_H * D_IN, D_IN, mlp_b + kk * MLP_H,
        nullptr, featb + kk * MLP_H, 4 * MLP_H, N_NODES, D_IN, 1);
  }
  bf16_gemm_kernel<<<dim3(MT, HID / 64), blk, 0, stream>>>(
      featb, 4 * MLP_H, wlint, 4 * MLP_H, mlp_lin_b, tmp, nullptr, HID, N_NODES, 4 * MLP_H, 0);
  bf16_t* combb = kbb;
  lsm_combine_kernel<<<dim3(N_NODES), blk, 0, stream>>>(tmp, x1, combb, N_NODES);

  // ---- final fc -> d_out [N, 64] ----
  bf16_gemm_kernel<<<dim3(MT, OUT_D / 64), blk, 0, stream>>>(
      combb, HID, wfct, HID, fc_b, out, nullptr, OUT_D, N_NODES, HID, 0);
}
